// MultiHeadAttention_65481071396388
// MI455X (gfx1250) — matrix operand order repaired and checked by reading, not compiled, not run on hardware
//
#include <hip/hip_runtime.h>

typedef __bf16 bf16;
typedef __attribute__((ext_vector_type(16))) __bf16 bf16x16;
typedef __attribute__((ext_vector_type(8)))  __bf16 bf16x8;
typedef __attribute__((ext_vector_type(4)))  __bf16 bf16x4;
typedef __attribute__((ext_vector_type(8)))  float  f32x8;
typedef __attribute__((ext_vector_type(4)))  unsigned int u32x4;
typedef __attribute__((ext_vector_type(8)))  int i32x8;
typedef __attribute__((ext_vector_type(4)))  int i32x4;

#define WMMA_BF16(a, b, c) \
  __builtin_amdgcn_wmma_f32_16x16x32_bf16(false, (a), false, (b), (short)0, (c), false, false)

static __device__ __forceinline__ bf16x16 pack16(bf16x8 lo, bf16x8 hi) {
  bf16x16 r;
#pragma unroll
  for (int i = 0; i < 8; ++i) { r[i] = lo[i]; r[i + 8] = hi[i]; }
  return r;
}

// ---------------------------------------------------------------- TDM issue --
// 2-D tile load via Tensor Data Mover (CDNA5 ISA).
static __device__ __forceinline__ void tdm_load_2d(const void* gptr, unsigned lds_off,
                                                   unsigned tensor_d0, unsigned tensor_d1,
                                                   unsigned tile_d0, unsigned tile_d1,
                                                   unsigned d0_stride) {
  unsigned long long ga = (unsigned long long)(size_t)gptr;
  u32x4 g0;
  g0[0] = 1u;                                                // count=1, user desc
  g0[1] = lds_off;                                           // LDS byte address
  g0[2] = (unsigned)(ga & 0xffffffffu);
  g0[3] = (unsigned)((ga >> 32) & 0x01ffffffu) | (2u << 30); // addr[56:32] | type=2
  i32x8 g1;
  g1[0] = (int)(1u << 16);                                   // data_size=1 -> 2 bytes
  g1[1] = (int)((tensor_d0 & 0xffffu) << 16);                // tensor_dim0[15:0]
  g1[2] = (int)((tensor_d0 >> 16) | ((tensor_d1 & 0xffffu) << 16));
  g1[3] = (int)((tensor_d1 >> 16) | (tile_d0 << 16));        // tile_dim0
  g1[4] = (int)tile_d1;                                      // tile_dim1 (tile_dim2=0)
  g1[5] = (int)d0_stride;                                    // tensor_dim0_stride[31:0]
  g1[6] = 0;
  g1[7] = 0;
  i32x4 z4 = {0, 0, 0, 0};
#if __has_include(<hip/amd_detail/amd_gfx1250_TDM.h>)
  i32x8 z8 = {0, 0, 0, 0, 0, 0, 0, 0};
  __builtin_amdgcn_tensor_load_to_lds(g0, g1, z4, z4, z8, 0);
#else
  __builtin_amdgcn_tensor_load_to_lds(g0, g1, z4, z4, 0);
#endif
}

// ---------------------------------------------------------------- convert ---
__global__ __launch_bounds__(256) void cvt_f32_to_bf16(const float* __restrict__ src,
                                                       bf16* __restrict__ dst, int n4) {
  int i = blockIdx.x * blockDim.x + threadIdx.x;
  if (i < n4) {
    float4 v = ((const float4*)src)[i];
    bf16x4 o;
    o[0] = (bf16)v.x; o[1] = (bf16)v.y; o[2] = (bf16)v.z; o[3] = (bf16)v.w;
    ((bf16x4*)dst)[i] = o;
  }
}

// ---------------------------------------------- TDM-staged GEMM-NT mainloop --
// C[128x64] per WG. A row-major [M,1024]; B row-major [N,1024] (out col = B row).
// Tiles staged by the Tensor Data Mover, double-buffered; 8 waves consume.
static __device__ __forceinline__ void gemm_mainloop_tdm(
    const bf16* __restrict__ A, const bf16* __restrict__ B,
    int m0, int f0, unsigned Mrows, unsigned Brows,
    bf16 (*sA)[128 * 32], bf16 (*sB)[64 * 32],
    int wave, int hi, int ln, f32x8 acc[4]) {
  if (wave == 0) {
    tdm_load_2d(A + (size_t)m0 * 1024, (unsigned)(size_t)&sA[0][0], 1024, Mrows, 32, 128, 1024);
    tdm_load_2d(B + (size_t)f0 * 1024, (unsigned)(size_t)&sB[0][0], 1024, Brows, 32, 64, 1024);
  }
  for (int ks = 0; ks < 32; ++ks) {
    const int cur = ks & 1;
    if (wave == 0) {
      if (ks + 1 < 32) {
        const int nxt = cur ^ 1;
        tdm_load_2d(A + (size_t)m0 * 1024 + (ks + 1) * 32, (unsigned)(size_t)&sA[nxt][0],
                    1024, Mrows, 32, 128, 1024);
        tdm_load_2d(B + (size_t)f0 * 1024 + (ks + 1) * 32, (unsigned)(size_t)&sB[nxt][0],
                    1024, Brows, 32, 64, 1024);
        __builtin_amdgcn_s_wait_tensorcnt((short)2);  // pair for `cur` has landed
      } else {
        __builtin_amdgcn_s_wait_tensorcnt((short)0);
      }
    }
    __syncthreads();  // tile `cur` visible to all waves
    // Batch all fragment loads, then the 4 WMMAs (partial dscnt waits).
    const bf16* Ar = &sA[cur][(wave * 16 + ln) * 32];
    bf16x16 a = pack16(*(const bf16x8*)(Ar + hi * 16),
                       *(const bf16x8*)(Ar + hi * 16 + 8));
    bf16x16 b[4];
#pragma unroll
    for (int ns = 0; ns < 4; ++ns) {
      const bf16* Br = &sB[cur][(ns * 16 + ln) * 32 + hi * 16];
      b[ns] = pack16(*(const bf16x8*)Br, *(const bf16x8*)(Br + 8));
    }
#pragma unroll
    for (int ns = 0; ns < 4; ++ns) acc[ns] = WMMA_BF16(a, b[ns], acc[ns]);
  }
}

// ------------------------------------------------------------------ QKV GEMM -
// C[m,f] = X[m,:].W[f,:] + bias[f];  m = b*2048+n, f = c*1024 + h*64 + d
// Q,K stored [c][bh][n][64] bf16; V stored transposed [bh][d][2048] bf16.
__global__ __launch_bounds__(256) void qkv_gemm_kernel(const bf16* __restrict__ X,
                                                       const bf16* __restrict__ W,
                                                       const float* __restrict__ bias,
                                                       bf16* __restrict__ qk,
                                                       bf16* __restrict__ vt) {
  __shared__ __align__(16) bf16 sA[2][128 * 32];
  __shared__ __align__(16) bf16 sB[2][64 * 32];
  const int lane = threadIdx.x & 31, wave = threadIdx.x >> 5;
  const int hi = lane >> 4, ln = lane & 15;
  const int m0 = blockIdx.x * 128;
  const int f0 = blockIdx.y * 64;

  f32x8 acc[4];
#pragma unroll
  for (int ns = 0; ns < 4; ++ns)
#pragma unroll
    for (int r = 0; r < 8; ++r) acc[ns][r] = 0.f;

  gemm_mainloop_tdm(X, W, m0, f0, 8192, 3072, sA, sB, wave, hi, ln, acc);

#pragma unroll
  for (int ns = 0; ns < 4; ++ns) {
    const int f = f0 + ns * 16 + ln;
    const int c = f >> 10, h = (f >> 6) & 15, d = f & 63;
    const float bv = bias[f];
#pragma unroll
    for (int r = 0; r < 8; ++r) {
      const int m = m0 + wave * 16 + r + 8 * hi;  // C layout row = r + 8*hi
      const int bb = m >> 11, nn = m & 2047;
      const int bh = bb * 16 + h;
      const bf16 val = (bf16)(acc[ns][r] + bv);
      if (c < 2) qk[(((size_t)c * 64 + bh) * 2048 + nn) * 64 + d] = val;
      else       vt[((size_t)bh * 64 + d) * 2048 + nn] = val;  // V transposed
    }
  }
}

// ------------------------------------------------------------ out projection -
__global__ __launch_bounds__(256) void out_gemm_kernel(const bf16* __restrict__ A,
                                                       const bf16* __restrict__ W,
                                                       const float* __restrict__ bias,
                                                       float* __restrict__ out) {
  __shared__ __align__(16) bf16 sA[2][128 * 32];
  __shared__ __align__(16) bf16 sB[2][64 * 32];
  const int lane = threadIdx.x & 31, wave = threadIdx.x >> 5;
  const int hi = lane >> 4, ln = lane & 15;
  const int m0 = blockIdx.x * 128;
  const int f0 = blockIdx.y * 64;

  f32x8 acc[4];
#pragma unroll
  for (int ns = 0; ns < 4; ++ns)
#pragma unroll
    for (int r = 0; r < 8; ++r) acc[ns][r] = 0.f;

  gemm_mainloop_tdm(A, W, m0, f0, 8192, 1024, sA, sB, wave, hi, ln, acc);

#pragma unroll
  for (int ns = 0; ns < 4; ++ns) {
    const int f = f0 + ns * 16 + ln;
    const float bv = bias[f];
#pragma unroll
    for (int r = 0; r < 8; ++r) {
      const int m = m0 + wave * 16 + r + 8 * hi;
      out[(size_t)m * 1024 + f] = acc[ns][r] + bv;
    }
  }
}

// ------------------------------------------------------- flash attention ----
// Per (b,h): S = Q K^T / 8, online softmax, O = P V. Wave owns 16 query rows.
// K and V^T tiles are WG-shared -> staged by TDM, double-buffered in LDS.
__global__ __launch_bounds__(256) void attn_kernel(const bf16* __restrict__ qk,
                                                   const bf16* __restrict__ vt,
                                                   bf16* __restrict__ ao) {
  __shared__ __align__(16) bf16 sK[2][32 * 64];      // [j][d] tile, 4KB each
  __shared__ __align__(16) bf16 sV[2][64 * 32];      // [d][j] tile, 4KB each
  __shared__ __align__(16) bf16 lds_p[8 * 16 * 32];  // per-wave 16x32 P tile
  const int lane = threadIdx.x & 31, wave = threadIdx.x >> 5;
  const int hi = lane >> 4, ln = lane & 15;
  const int bh = blockIdx.y;
  const int q0 = blockIdx.x * 128 + wave * 16;

  const bf16* Q  = qk + (size_t)bh * 2048 * 64;
  const bf16* Kp = qk + (size_t)(64 + bh) * 2048 * 64;
  const bf16* Vt = vt + (size_t)bh * 64 * 2048;

  // Preload Q A-frags for the whole K=64 reduction (2 k-steps of 32)
  bf16x16 aq[2];
  {
    const bf16* Qrow = Q + (size_t)(q0 + ln) * 64;
#pragma unroll
    for (int kt = 0; kt < 2; ++kt)
      aq[kt] = pack16(*(const bf16x8*)(Qrow + kt * 32 + hi * 16),
                      *(const bf16x8*)(Qrow + kt * 32 + hi * 16 + 8));
  }

  f32x8 o[4];
  float mrow[8], lrow[8];
#pragma unroll
  for (int r = 0; r < 8; ++r) {
    mrow[r] = -1e30f; lrow[r] = 0.f;
#pragma unroll
    for (int dg = 0; dg < 4; ++dg) o[dg][r] = 0.f;
  }

  bf16* myp = lds_p + wave * 512;

  if (wave == 0) {
    tdm_load_2d(Kp, (unsigned)(size_t)&sK[0][0], 64, 2048, 64, 32, 64);
    tdm_load_2d(Vt, (unsigned)(size_t)&sV[0][0], 2048, 64, 32, 64, 2048);
  }

  for (int js = 0; js < 64; ++js) {      // j-block = js*32 keys
    const int cur = js & 1;
    if (wave == 0) {
      if (js + 1 < 64) {
        const int nxt = cur ^ 1;
        tdm_load_2d(Kp + (size_t)(js + 1) * 32 * 64, (unsigned)(size_t)&sK[nxt][0],
                    64, 2048, 64, 32, 64);
        tdm_load_2d(Vt + (size_t)(js + 1) * 32, (unsigned)(size_t)&sV[nxt][0],
                    2048, 64, 32, 64, 2048);
        __builtin_amdgcn_s_wait_tensorcnt((short)2);
      } else {
        __builtin_amdgcn_s_wait_tensorcnt((short)0);
      }
    }
    __syncthreads();

    // ---- scores for 32 keys: batch K fragments then 4 WMMAs ----
    bf16x16 bk[2][2];
#pragma unroll
    for (int jt = 0; jt < 2; ++jt) {
      const bf16* Krow = &sK[cur][(jt * 16 + ln) * 64];
#pragma unroll
      for (int kt = 0; kt < 2; ++kt)
        bk[jt][kt] = pack16(*(const bf16x8*)(Krow + kt * 32 + hi * 16),
                            *(const bf16x8*)(Krow + kt * 32 + hi * 16 + 8));
    }
    f32x8 s[2];
#pragma unroll
    for (int jt = 0; jt < 2; ++jt) {
      f32x8 sa;
#pragma unroll
      for (int r = 0; r < 8; ++r) sa[r] = 0.f;
      sa = WMMA_BF16(aq[0], bk[jt][0], sa);
      sa = WMMA_BF16(aq[1], bk[jt][1], sa);
      s[jt] = sa;
    }

    // ---- online softmax: rows live in 16-lane halves of the C layout ----
#pragma unroll
    for (int r = 0; r < 8; ++r) {
      float v0 = s[0][r] * 0.125f;  // 1/sqrt(64)
      float v1 = s[1][r] * 0.125f;
      float v = fmaxf(v0, v1);
      v = fmaxf(v, __shfl_xor(v, 1));
      v = fmaxf(v, __shfl_xor(v, 2));
      v = fmaxf(v, __shfl_xor(v, 4));
      v = fmaxf(v, __shfl_xor(v, 8));
      const float mnew  = fmaxf(mrow[r], v);
      const float alpha = __expf(mrow[r] - mnew);
      const float p0 = __expf(v0 - mnew);
      const float p1 = __expf(v1 - mnew);
      float ps = p0 + p1;
      ps += __shfl_xor(ps, 1);
      ps += __shfl_xor(ps, 2);
      ps += __shfl_xor(ps, 4);
      ps += __shfl_xor(ps, 8);
      lrow[r] = lrow[r] * alpha + ps;
      mrow[r] = mnew;
#pragma unroll
      for (int dg = 0; dg < 4; ++dg) o[dg][r] *= alpha;
      myp[(r + 8 * hi) * 32 + ln]      = (bf16)p0;  // stage P for lane-transpose
      myp[(r + 8 * hi) * 32 + 16 + ln] = (bf16)p1;
    }

    // ---- O += P(16x32) x V(32x64): A from LDS, B from staged V^T ----
    bf16x16 ap = pack16(*(const bf16x8*)(myp + ln * 32 + hi * 16),
                        *(const bf16x8*)(myp + ln * 32 + hi * 16 + 8));
    bf16x16 bv[4];
#pragma unroll
    for (int dg = 0; dg < 4; ++dg) {
      const bf16* vr = &sV[cur][(dg * 16 + ln) * 32 + hi * 16];
      bv[dg] = pack16(*(const bf16x8*)vr, *(const bf16x8*)(vr + 8));
    }
#pragma unroll
    for (int dg = 0; dg < 4; ++dg) o[dg] = WMMA_BF16(ap, bv[dg], o[dg]);
  }

  // ---- normalize + store attn output [b*2048+n][h*64+d] bf16 ----
  const int bb = bh >> 4, h = bh & 15;
#pragma unroll
  for (int dg = 0; dg < 4; ++dg)
#pragma unroll
    for (int r = 0; r < 8; ++r) {
      const int row = q0 + r + 8 * hi;
      const float v = o[dg][r] / lrow[r];
      ao[((size_t)(bb * 2048 + row)) * 1024 + h * 64 + dg * 16 + ln] = (bf16)v;
    }
}

// ------------------------------------------------------------------- launch --
extern "C" void kernel_launch(void* const* d_in, const int* in_sizes, int n_in,
                              void* d_out, int out_size, void* d_ws, size_t ws_size,
                              hipStream_t stream) {
  (void)in_sizes; (void)n_in; (void)out_size; (void)ws_size;
  const float* x     = (const float*)d_in[0];  // [4,2048,1024]
  const float* qkv_w = (const float*)d_in[1];  // [3072,1024]
  const float* qkv_b = (const float*)d_in[2];  // [3072]
  const float* out_w = (const float*)d_in[3];  // [1024,1024]
  const float* out_b = (const float*)d_in[4];  // [1024]
  float* out = (float*)d_out;

  bf16* p = (bf16*)d_ws;
  bf16* xb   = p;  p += (size_t)8192 * 1024;         // x in bf16
  bf16* wqkv = p;  p += (size_t)3072 * 1024;         // qkv_w bf16
  bf16* wo   = p;  p += (size_t)1024 * 1024;         // out_w bf16
  bf16* qk   = p;  p += (size_t)2 * 64 * 2048 * 64;  // Q,K [c][bh][n][64]
  bf16* vt   = p;  p += (size_t)64 * 64 * 2048;      // V^T [bh][d][n]
  bf16* ao   = p;  p += (size_t)8192 * 1024;         // attn out bf16

  cvt_f32_to_bf16<<<(8192 * 1024 / 4) / 256, 256, 0, stream>>>(x, xb, 8192 * 1024 / 4);
  cvt_f32_to_bf16<<<(3072 * 1024 / 4) / 256, 256, 0, stream>>>(qkv_w, wqkv, 3072 * 1024 / 4);
  cvt_f32_to_bf16<<<(1024 * 1024 / 4) / 256, 256, 0, stream>>>(out_w, wo, 1024 * 1024 / 4);

  qkv_gemm_kernel<<<dim3(64, 48), 256, 0, stream>>>(xb, wqkv, qkv_b, qk, vt);
  attn_kernel    <<<dim3(16, 64), 256, 0, stream>>>(qk, vt, ao);
  out_gemm_kernel<<<dim3(64, 16), 256, 0, stream>>>(ao, wo, out_b, out);
}
